// GIN_GNN_58737972740393
// MI455X (gfx1250) — matrix operand order repaired and checked by reading, not compiled, not run on hardware
//
#include <hip/hip_runtime.h>
#include <hip/hip_bf16.h>
#include <math.h>

typedef __attribute__((ext_vector_type(16))) _Float16 v16h;
typedef __attribute__((ext_vector_type(8)))  _Float16 v8h;
typedef __attribute__((ext_vector_type(8)))  float    v8f;

#define CDIV(a,b) (((a)+(b)-1)/(b))

static __device__ __forceinline__ float elu01f(float x) {
    return x > 0.f ? x : 0.1f * expm1f(x);
}

static __device__ __forceinline__ void atomicMaxF(float* addr, float val) {
    int* ai = (int*)addr;
    int old = __float_as_int(*addr);
    while (__int_as_float(old) < val) {
        int assumed = old;
        old = atomicCAS(ai, assumed, __float_as_int(val));
        if (old == assumed) break;
    }
}

// ---- pack x0 = concat(h0, coord0) into [N,32], zero-padded cols 19..31 ----
__global__ void gin_pack_x0(const float* __restrict__ h0, const float* __restrict__ coord,
                            float* __restrict__ X, int n) {
    int idx = blockIdx.x * blockDim.x + threadIdx.x;
    if (idx >= n * 32) return;
    int r = idx >> 5, c = idx & 31;
    float v = 0.f;
    if (c < 16) v = h0[r * 16 + c];
    else if (c < 19) v = coord[r * 3 + (c - 16)];
    X[idx] = v;
}

// ---- convert weight [K,128] f32 -> transposed f16 [128,Kpad] (zero-pad K) ----
__global__ void gin_convert_wt(const float* __restrict__ w, _Float16* __restrict__ wt,
                               int K, int Kpad) {
    int idx = blockIdx.x * blockDim.x + threadIdx.x;
    if (idx >= 128 * Kpad) return;
    int j = idx / Kpad, k = idx - j * Kpad;
    wt[idx] = (k < K) ? (_Float16)w[k * 128 + j] : (_Float16)0.f;
}

// ---- edge scatter-add: agg[dst] += x[src], dim is a power of two ----
__global__ void gin_scatter_add(const float* __restrict__ X, const int* __restrict__ src,
                                const int* __restrict__ dst, float* __restrict__ agg,
                                int total, int lg) {
    int idx = blockIdx.x * blockDim.x + threadIdx.x;
    if (idx >= total) return;
    int e = idx >> lg;
    int c = idx & ((1 << lg) - 1);
    atomicAdd(&agg[((size_t)dst[e] << lg) + c], X[((size_t)src[e] << lg) + c]);
}

// ---- A16 = f16(X + AGG) ----
__global__ void gin_prep_a16(const float* __restrict__ X, const float* __restrict__ agg,
                             _Float16* __restrict__ A16, int total) {
    int idx = blockIdx.x * blockDim.x + threadIdx.x;
    if (idx >= total) return;
    A16[idx] = (_Float16)(X[idx] + agg[idx]);
}

// ---- WMMA GEMM: out[N,128] = A[N,K] * Bt[128,K]^T + bias (ACT=1 -> elu01) ----
// Each wave computes a 32x16 tile; 8 waves/block cover 32 rows x 128 cols.
template <int ACT>
__global__ __launch_bounds__(256) void gin_gemm_wmma(
        const _Float16* __restrict__ A, const _Float16* __restrict__ Bt,
        const float* __restrict__ bias, float* __restrict__ out, int K) {
    const int lane  = threadIdx.x & 31;
    const int wave  = threadIdx.x >> 5;
    const int r     = lane & 15;
    const int hi    = lane >> 4;
    const int n0    = wave << 4;
    const int rbase = blockIdx.x << 5;
    const _Float16* a0p = A + (size_t)(rbase + r) * K;
    const _Float16* a1p = A + (size_t)(rbase + 16 + r) * K;
    const _Float16* bp  = Bt + (size_t)(n0 + r) * K;
    v8f c0 = {}; v8f c1 = {};
    for (int k0 = 0; k0 < K; k0 += 32) {
        // B fragment: lane holds col n0+r, K = {k0+hi*8..+7, k0+16+hi*8..+7}
        v8h blo = *(const v8h*)(bp + k0 + hi * 8);
        v8h bhi = *(const v8h*)(bp + k0 + 16 + hi * 8);
        v16h b = __builtin_shufflevector(blo, bhi, 0,1,2,3,4,5,6,7,8,9,10,11,12,13,14,15);
        // A fragment: lane holds row, K = {k0+hi*8..+7, k0+16+hi*8..+7}
        v8h a0lo = *(const v8h*)(a0p + k0 + hi * 8);
        v8h a0hi = *(const v8h*)(a0p + k0 + 16 + hi * 8);
        v8h a1lo = *(const v8h*)(a1p + k0 + hi * 8);
        v8h a1hi = *(const v8h*)(a1p + k0 + 16 + hi * 8);
        v16h a0 = __builtin_shufflevector(a0lo, a0hi, 0,1,2,3,4,5,6,7,8,9,10,11,12,13,14,15);
        v16h a1 = __builtin_shufflevector(a1lo, a1hi, 0,1,2,3,4,5,6,7,8,9,10,11,12,13,14,15);
        c0 = __builtin_amdgcn_wmma_f32_16x16x32_f16(false, a0, false, b, (short)0, c0, false, false);
        c1 = __builtin_amdgcn_wmma_f32_16x16x32_f16(false, a1, false, b, (short)0, c1, false, false);
    }
    const int col = n0 + r;
    const float bs = bias[col];
#pragma unroll
    for (int v = 0; v < 8; ++v) {
        int m = v + (hi << 3);
        float x0 = c0[v] + bs;
        float x1 = c1[v] + bs;
        if (ACT) { x0 = elu01f(x0); x1 = elu01f(x1); }
        out[(size_t)(rbase + m) * 128 + col] = x0;
        out[(size_t)(rbase + 16 + m) * 128 + col] = x1;
    }
}

// ---- BN column statistics (partial, atomically combined) ----
__global__ void gin_bn_stats(const float* __restrict__ h, float* __restrict__ sum,
                             float* __restrict__ sumsq, int n) {
    int c = threadIdx.x; // blockDim.x == 128
    float s = 0.f, s2 = 0.f;
    for (int r = blockIdx.x; r < n; r += gridDim.x) {
        float v = h[(size_t)r * 128 + c];
        s += v; s2 += v * v;
    }
    atomicAdd(&sum[c], s);
    atomicAdd(&sumsq[c], s2);
}

__global__ void gin_bn_finalize(const float* __restrict__ sum, const float* __restrict__ sumsq,
                                const float* __restrict__ gm, const float* __restrict__ bt,
                                float* __restrict__ scale, float* __restrict__ shift, int n) {
    int c = threadIdx.x;
    if (c >= 128) return;
    float inv_n = 1.f / (float)n;
    float mean = sum[c] * inv_n;
    float var  = fmaxf(sumsq[c] * inv_n - mean * mean, 0.f);
    float sc   = gm[c] * rsqrtf(var + 1e-5f);
    scale[c] = sc;
    shift[c] = bt[c] - mean * sc;
}

// ---- apply BN + ReLU, emit f16 for next GEMM ----
__global__ void gin_bn_relu_f16(const float* __restrict__ h, const float* __restrict__ scale,
                                const float* __restrict__ shift, _Float16* __restrict__ A16,
                                int total) {
    int idx = blockIdx.x * blockDim.x + threadIdx.x;
    if (idx >= total) return;
    int c = idx & 127;
    float v = fmaxf(h[idx] * scale[c] + shift[c], 0.f);
    A16[idx] = (_Float16)v;
}

// ---- pooling ----
__global__ void gin_pool_init(float* __restrict__ pmax, int total) {
    int idx = blockIdx.x * blockDim.x + threadIdx.x;
    if (idx < total) pmax[idx] = -1e30f;
}

__global__ void gin_pool_accum(const float* __restrict__ X, const int* __restrict__ batch,
                               float* __restrict__ psum, float* __restrict__ pmax, int total) {
    int idx = blockIdx.x * blockDim.x + threadIdx.x;
    if (idx >= total) return;
    int r = idx >> 7, c = idx & 127;
    int b = batch[r];
    float v = X[idx];
    atomicAdd(&psum[b * 128 + c], v);
    atomicMaxF(&pmax[b * 128 + c], v);
}

__global__ void gin_pool_count(const int* __restrict__ batch, float* __restrict__ cnt, int n) {
    int idx = blockIdx.x * blockDim.x + threadIdx.x;
    if (idx < n) atomicAdd(&cnt[batch[idx]], 1.f);
}

// ---- classifier layer 1: z1 = elu01([mean|max|g0] @ cw1 + cb1), B=64 ----
__global__ void gin_cls1(const float* __restrict__ psum, const float* __restrict__ pmax,
                         const float* __restrict__ cnt, const float* __restrict__ g0,
                         const float* __restrict__ cw1, const float* __restrict__ cb1,
                         float* __restrict__ z1) {
    int idx = blockIdx.x * blockDim.x + threadIdx.x;
    if (idx >= 64 * 128) return;
    int b = idx >> 7, j = idx & 127;
    float inv = 1.f / fmaxf(cnt[b], 1.f);
    float acc = cb1[j];
    for (int k = 0; k < 128; ++k) acc += psum[b * 128 + k] * inv * cw1[k * 128 + j];
    for (int k = 0; k < 128; ++k) acc += pmax[b * 128 + k] * cw1[(128 + k) * 128 + j];
    for (int k = 0; k < 8;  ++k) acc += g0[b * 8 + k] * cw1[(256 + k) * 128 + j];
    z1[idx] = elu01f(acc);
}

// ---- classifier BN over B=64 rows (in place, one thread owns one column) ----
__global__ void gin_cls_bn(float* __restrict__ z1, const float* __restrict__ gm,
                           const float* __restrict__ bt) {
    int j = threadIdx.x;
    if (j >= 128) return;
    float s = 0.f, s2 = 0.f;
    for (int b = 0; b < 64; ++b) { float v = z1[b * 128 + j]; s += v; s2 += v * v; }
    float mean = s * (1.f / 64.f);
    float var  = fmaxf(s2 * (1.f / 64.f) - mean * mean, 0.f);
    float sc   = gm[j] * rsqrtf(var + 1e-5f);
    float sh   = bt[j] - mean * sc;
    for (int b = 0; b < 64; ++b) z1[b * 128 + j] = z1[b * 128 + j] * sc + sh;
}

// ---- classifier layer 2 + softmax ----
__global__ void gin_cls2(const float* __restrict__ z1, const float* __restrict__ cw2,
                         const float* __restrict__ cb2, float* __restrict__ out) {
    int b = blockIdx.x * blockDim.x + threadIdx.x;
    if (b >= 64) return;
    float l0 = cb2[0], l1 = cb2[1];
    for (int k = 0; k < 128; ++k) {
        float v = z1[b * 128 + k];
        l0 += v * cw2[k * 2 + 0];
        l1 += v * cw2[k * 2 + 1];
    }
    float m = fmaxf(l0, l1);
    float e0 = __expf(l0 - m), e1 = __expf(l1 - m);
    float inv = 1.f / (e0 + e1);
    out[b * 2 + 0] = e0 * inv;
    out[b * 2 + 1] = e1 * inv;
}

extern "C" void kernel_launch(void* const* d_in, const int* in_sizes, int n_in,
                              void* d_out, int out_size, void* d_ws, size_t ws_size,
                              hipStream_t stream) {
    (void)in_sizes; (void)n_in; (void)out_size; (void)ws_size;
    const int N = 100000, E = 1600000;

    const float* h0     = (const float*)d_in[0];
    const float* coord0 = (const float*)d_in[1];
    const float* g0     = (const float*)d_in[2];
    const int*   edge   = (const int*)d_in[3];
    const int*   src    = edge;
    const int*   dst    = edge + E;
    const int*   batch  = (const int*)d_in[4];
    const float* w1_0 = (const float*)d_in[5];
    const float* b1_0 = (const float*)d_in[6];
    const float* gm_0 = (const float*)d_in[7];
    const float* bt_0 = (const float*)d_in[8];
    const float* w2_0 = (const float*)d_in[9];
    const float* b2_0 = (const float*)d_in[10];
    const float* w1_r = (const float*)d_in[11];
    const float* b1_r = (const float*)d_in[12];
    const float* gm_r = (const float*)d_in[13];
    const float* bt_r = (const float*)d_in[14];
    const float* w2_r = (const float*)d_in[15];
    const float* b2_r = (const float*)d_in[16];
    const float* cw1  = (const float*)d_in[17];
    const float* cb1  = (const float*)d_in[18];
    const float* cgm  = (const float*)d_in[19];
    const float* cbt  = (const float*)d_in[20];
    const float* cw2  = (const float*)d_in[21];
    const float* cb2  = (const float*)d_in[22];

    // workspace carve-out (256B aligned)
    char* ws = (char*)d_ws;
    size_t off = 0;
    auto take = [&](size_t bytes) -> char* {
        char* p = ws + off;
        off = (off + bytes + 255) & ~(size_t)255;
        return p;
    };
    float*    X     = (float*)take((size_t)N * 128 * 4);   // activations (layer0: [N,32] slice)
    float*    AGG   = (float*)take((size_t)N * 128 * 4);   // aggregation, reused as pre-BN GEMM1 out
    _Float16* A16   = (_Float16*)take((size_t)N * 128 * 2);// f16 GEMM input (reused for BN output)
    _Float16* WT1_0 = (_Float16*)take(128 * 32 * 2);
    _Float16* WT2_0 = (_Float16*)take(128 * 128 * 2);
    _Float16* WT1R  = (_Float16*)take(2 * 128 * 128 * 2);
    _Float16* WT2R  = (_Float16*)take(2 * 128 * 128 * 2);
    float*    SUM   = (float*)take(128 * 4);
    float*    SUMSQ = (float*)take(128 * 4);
    float*    SCALE = (float*)take(128 * 4);
    float*    SHIFT = (float*)take(128 * 4);
    float*    PSUM  = (float*)take(64 * 128 * 4);
    float*    PMAX  = (float*)take(64 * 128 * 4);
    float*    CNT   = (float*)take(64 * 4);
    float*    Z1    = (float*)take(64 * 128 * 4);

    // weight conversion (f32 [K,128] -> f16 transposed [128,Kpad])
    gin_convert_wt<<<CDIV(128 * 32, 256), 256, 0, stream>>>(w1_0, WT1_0, 19, 32);
    gin_convert_wt<<<CDIV(128 * 128, 256), 256, 0, stream>>>(w2_0, WT2_0, 128, 128);
    for (int i = 0; i < 2; ++i) {
        gin_convert_wt<<<CDIV(128 * 128, 256), 256, 0, stream>>>(
            w1_r + (size_t)i * 128 * 128, WT1R + (size_t)i * 128 * 128, 128, 128);
        gin_convert_wt<<<CDIV(128 * 128, 256), 256, 0, stream>>>(
            w2_r + (size_t)i * 128 * 128, WT2R + (size_t)i * 128 * 128, 128, 128);
    }

    // x0 = concat(h0, coord0), padded to 32 cols
    gin_pack_x0<<<CDIV(N * 32, 256), 256, 0, stream>>>(h0, coord0, X, N);

    const int             Ks[3]  = {32, 128, 128};
    const int             lgs[3] = {5, 7, 7};
    const _Float16* const wt1s[3] = {WT1_0, WT1R, WT1R + 128 * 128};
    const _Float16* const wt2s[3] = {WT2_0, WT2R, WT2R + 128 * 128};
    const float* const    b1s[3]  = {b1_0, b1_r, b1_r + 128};
    const float* const    b2s[3]  = {b2_0, b2_r, b2_r + 128};
    const float* const    gms[3]  = {gm_0, gm_r, gm_r + 128};
    const float* const    bts[3]  = {bt_0, bt_r, bt_r + 128};

    for (int L = 0; L < 3; ++L) {
        const int K = Ks[L];
        // 1. aggregate neighbors
        hipMemsetAsync(AGG, 0, (size_t)N * K * 4, stream);
        gin_scatter_add<<<CDIV(E * K, 256), 256, 0, stream>>>(X, src, dst, AGG, E * K, lgs[L]);
        // 2. A16 = f16(x + agg)
        gin_prep_a16<<<CDIV(N * K, 256), 256, 0, stream>>>(X, AGG, A16, N * K);
        // 3. GEMM1 -> pre-BN activations (reuse AGG buffer)
        gin_gemm_wmma<0><<<N / 32, 256, 0, stream>>>(A16, wt1s[L], b1s[L], AGG, K);
        // 4. BatchNorm statistics + ReLU, emit f16
        hipMemsetAsync(SUM, 0, 128 * 4, stream);
        hipMemsetAsync(SUMSQ, 0, 128 * 4, stream);
        gin_bn_stats<<<512, 128, 0, stream>>>(AGG, SUM, SUMSQ, N);
        gin_bn_finalize<<<1, 128, 0, stream>>>(SUM, SUMSQ, gms[L], bts[L], SCALE, SHIFT, N);
        gin_bn_relu_f16<<<CDIV(N * 128, 256), 256, 0, stream>>>(AGG, SCALE, SHIFT, A16, N * 128);
        // 5. GEMM2 + ELU(0.1) -> new activations
        gin_gemm_wmma<1><<<N / 32, 256, 0, stream>>>(A16, wt2s[L], b2s[L], X, 128);
    }

    // pooling (mean + max per graph)
    hipMemsetAsync(PSUM, 0, 64 * 128 * 4, stream);
    hipMemsetAsync(CNT, 0, 64 * 4, stream);
    gin_pool_init<<<CDIV(64 * 128, 256), 256, 0, stream>>>(PMAX, 64 * 128);
    gin_pool_accum<<<CDIV(N * 128, 256), 256, 0, stream>>>(X, batch, PSUM, PMAX, N * 128);
    gin_pool_count<<<CDIV(N, 256), 256, 0, stream>>>(batch, CNT, N);

    // classifier
    gin_cls1<<<CDIV(64 * 128, 256), 256, 0, stream>>>(PSUM, PMAX, CNT, g0, cw1, cb1, Z1);
    gin_cls_bn<<<1, 128, 0, stream>>>(Z1, cgm, cbt);
    gin_cls2<<<1, 64, 0, stream>>>(Z1, cw2, cb2, (float*)d_out);
}
